// GNNNodeEncoder_43714177138808
// MI455X (gfx1250) — matrix operand order repaired and checked by reading, not compiled, not run on hardware
//
#include <hip/hip_runtime.h>

// ---------------- constants ----------------
static constexpr int EMB   = 128;   // d
static constexpr int HID   = 256;   // 2d
static constexpr int L     = 5;
static constexpr float BNEPS = 1e-5f;

typedef __attribute__((ext_vector_type(16))) _Float16 v16h;
typedef __attribute__((ext_vector_type(8)))  _Float16 v8h;
typedef __attribute__((ext_vector_type(8)))  float    v8f;

// ---------------- input node embedding: h = atom[x0] + chir[x1] + hyb[x2] ----------------
__global__ void node_embed_kernel(const int* __restrict__ x,
                                  const float* __restrict__ atom_emb,
                                  const float* __restrict__ chir_emb,
                                  const float* __restrict__ hyb_emb,
                                  float* __restrict__ h, int n_nodes) {
  int tid = blockIdx.x * blockDim.x + threadIdx.x;
  if (tid >= n_nodes * EMB) return;
  int n = tid >> 7;          // /128
  int c = tid & (EMB - 1);
  int a  = x[n * 3 + 0];
  int ch = x[n * 3 + 1];
  int hy = x[n * 3 + 2];
  h[tid] = atom_emb[a * EMB + c] + chir_emb[ch * EMB + c] + hyb_emb[hy * EMB + c];
}

// ---------------- one-shot: f32 weights -> f16, transposed (B columns contiguous) ----------------
// W1t[l][n][k] = (f16) W1[l][k][n]   (n in [0,256), k in [0,128))
// W2t[l][n][k] = (f16) W2[l][k][n]   (n in [0,128), k in [0,256))
__global__ void convert_weights_kernel(const float* __restrict__ W1,
                                       const float* __restrict__ W2,
                                       _Float16* __restrict__ W1t,
                                       _Float16* __restrict__ W2t) {
  int tid = blockIdx.x * blockDim.x + threadIdx.x;
  const int per = L * EMB * HID;  // 163840
  if (tid < per) {
    int l = tid / (EMB * HID);
    int r = tid % (EMB * HID);
    int n = r / EMB;            // 0..255
    int k = r % EMB;            // 0..127
    W1t[tid] = (_Float16)W1[(size_t)l * EMB * HID + (size_t)k * HID + n];
  } else if (tid < 2 * per) {
    int t = tid - per;
    int l = t / (EMB * HID);
    int r = t % (EMB * HID);
    int n = r / HID;            // 0..127
    int k = r % HID;            // 0..255
    W2t[t] = (_Float16)W2[(size_t)l * EMB * HID + (size_t)k * EMB + n];
  }
}

// ---------------- per-edge: msg = h[src] + e1[ea0] + e2[ea1]; atomic scatter-add to agg[dst] ----
// 32 lanes per edge, 4 channels (float4) per lane.
__global__ void edge_scatter_kernel(const float* __restrict__ h,
                                    const int* __restrict__ ei,   // [2,E]
                                    const int* __restrict__ ea,   // [E,2]
                                    const float* __restrict__ e1l,
                                    const float* __restrict__ e2l,
                                    float* __restrict__ agg,
                                    int n_edges) {
  int tid = blockIdx.x * blockDim.x + threadIdx.x;
  int e = tid >> 5;
  if (e >= n_edges) return;
  int c4 = (tid & 31) << 2;
  int s  = ei[e];
  int d  = ei[n_edges + e];
  int a0 = ea[2 * e];
  int a1 = ea[2 * e + 1];
  const float4 hq = *(const float4*)(h   + (size_t)s  * EMB + c4);
  const float4 q1 = *(const float4*)(e1l + (size_t)a0 * EMB + c4);
  const float4 q2 = *(const float4*)(e2l + (size_t)a1 * EMB + c4);
  float m0 = hq.x + q1.x + q2.x;
  float m1 = hq.y + q1.y + q2.y;
  float m2 = hq.z + q1.z + q2.z;
  float m3 = hq.w + q1.w + q2.w;
  float* o = agg + (size_t)d * EMB + c4;
  atomicAdd(o + 0, m0);
  atomicAdd(o + 1, m1);
  atomicAdd(o + 2, m2);
  atomicAdd(o + 3, m3);
}

// ---------------- GEMM1: h2a = relu(agg[N,128] @ W1 + b1) -> f16 [N,256] ----------------
// one wave = one 16x64 output tile (4 D fragments); A fragment loaded once per
// K-chunk and reused across 4 WMMAs. K=128 -> 4 chunks -> 16 WMMA per wave.
__global__ void gemm1_relu_kernel(const float* __restrict__ A,       // agg [N,128] f32
                                  const _Float16* __restrict__ Bt,   // W1t[l] [256][128]
                                  const float* __restrict__ bias,    // b1[l] [256]
                                  _Float16* __restrict__ Cout,       // h2a [N,256] f16
                                  int n_nodes) {
  const int lane = threadIdx.x & 31;
  const int wave = threadIdx.x >> 5;
  const int w = blockIdx.x * (blockDim.x >> 5) + wave;
  const int mt = (n_nodes + 15) >> 4;
  if (w >= mt * (HID / 64)) return;          // 4 column groups of 64
  const int tm = w >> 2;
  const int tg = w & 3;                      // column group

  const int mrow = tm * 16 + (lane & 15);
  const int mr   = (mrow < n_nodes) ? mrow : (n_nodes - 1);
  const int kba  = (lane < 16) ? 0 : 16;     // A fragment K base (contiguous 16-k per lane half)
  const int nc0  = tg * 64 + (lane & 15);    // column of fragment j is nc0 + 16*j
  const int kbb  = (lane < 16) ? 0 : 16;     // B fragment K base

  v8f acc[4] = {};
#pragma unroll
  for (int kk = 0; kk < EMB; kk += 32) {
    // ---- A fragment (f32 -> f16 on the fly) ----
    const float* ap = A + (size_t)mr * EMB + kk + kba;
    const float4 l0 = *(const float4*)(ap + 0);
    const float4 l1 = *(const float4*)(ap + 4);
    const float4 h0 = *(const float4*)(ap + 8);
    const float4 h1 = *(const float4*)(ap + 12);
    v16h af;
    af[0]  = (_Float16)l0.x; af[1]  = (_Float16)l0.y; af[2]  = (_Float16)l0.z; af[3]  = (_Float16)l0.w;
    af[4]  = (_Float16)l1.x; af[5]  = (_Float16)l1.y; af[6]  = (_Float16)l1.z; af[7]  = (_Float16)l1.w;
    af[8]  = (_Float16)h0.x; af[9]  = (_Float16)h0.y; af[10] = (_Float16)h0.z; af[11] = (_Float16)h0.w;
    af[12] = (_Float16)h1.x; af[13] = (_Float16)h1.y; af[14] = (_Float16)h1.z; af[15] = (_Float16)h1.w;
    // ---- 4 B fragments, 4 WMMAs reusing af ----
#pragma unroll
    for (int j = 0; j < 4; ++j) {
      const v16h bf = *(const v16h*)(Bt + (size_t)(nc0 + 16 * j) * EMB + kk + kbb);
      acc[j] = __builtin_amdgcn_wmma_f32_16x16x32_f16(false, af, false, bf,
                                                      (short)0, acc[j], false, false);
    }
  }

  const int rbase = tm * 16 + ((lane < 16) ? 0 : 8);
#pragma unroll
  for (int j = 0; j < 4; ++j) {
    const int ncol = nc0 + 16 * j;
    const float bv = bias[ncol];
#pragma unroll
    for (int r = 0; r < 8; ++r) {
      int row = rbase + r;
      if (row < n_nodes) {
        float v = acc[j][r] + bv;
        v = v > 0.f ? v : 0.f;     // ReLU
        Cout[(size_t)row * HID + ncol] = (_Float16)v;
      }
    }
  }
}

// ---------------- GEMM2: h3 = h2a[N,256] @ W2 + b2 -> f32 [N,128] ----------------
// one wave = 16x64 tile; K=256 -> 8 chunks -> 32 WMMA per wave.
__global__ void gemm2_kernel(const _Float16* __restrict__ A,     // h2a [N,256] f16
                             const _Float16* __restrict__ Bt,    // W2t[l] [128][256]
                             const float* __restrict__ bias,     // b2[l] [128]
                             float* __restrict__ Cout,           // h3 [N,128] f32
                             int n_nodes) {
  const int lane = threadIdx.x & 31;
  const int wave = threadIdx.x >> 5;
  const int w = blockIdx.x * (blockDim.x >> 5) + wave;
  const int mt = (n_nodes + 15) >> 4;
  if (w >= mt * (EMB / 64)) return;          // 2 column groups of 64
  const int tm = w >> 1;
  const int tg = w & 1;

  const int mrow = tm * 16 + (lane & 15);
  const int mr   = (mrow < n_nodes) ? mrow : (n_nodes - 1);
  const int kba  = (lane < 16) ? 0 : 16;
  const int nc0  = tg * 64 + (lane & 15);
  const int kbb  = (lane < 16) ? 0 : 16;

  v8f acc[4] = {};
#pragma unroll
  for (int kk = 0; kk < HID; kk += 32) {
    const _Float16* ap = A + (size_t)mr * HID + kk + kba;
    const v8h lo = *(const v8h*)(ap);
    const v8h hi = *(const v8h*)(ap + 8);
    v16h af;
#pragma unroll
    for (int j = 0; j < 8; ++j) { af[j] = lo[j]; af[8 + j] = hi[j]; }
#pragma unroll
    for (int j = 0; j < 4; ++j) {
      const v16h bf = *(const v16h*)(Bt + (size_t)(nc0 + 16 * j) * HID + kk + kbb);
      acc[j] = __builtin_amdgcn_wmma_f32_16x16x32_f16(false, af, false, bf,
                                                      (short)0, acc[j], false, false);
    }
  }

  const int rbase = tm * 16 + ((lane < 16) ? 0 : 8);
#pragma unroll
  for (int j = 0; j < 4; ++j) {
    const int ncol = nc0 + 16 * j;
    const float bv = bias[ncol];
#pragma unroll
    for (int r = 0; r < 8; ++r) {
      int row = rbase + r;
      if (row < n_nodes) Cout[(size_t)row * EMB + ncol] = acc[j][r] + bv;
    }
  }
}

// ---------------- BN stats: per-channel sum and sum of squares over all rows ----------------
__global__ void bn_stats_kernel(const float* __restrict__ h3,
                                float* __restrict__ sums,
                                float* __restrict__ sumsq,
                                int n_nodes) {
  const int c = threadIdx.x;         // blockDim.x == 128 channels
  float s = 0.f, s2 = 0.f;
  for (int r = blockIdx.x; r < n_nodes; r += gridDim.x) {
    float v = h3[(size_t)r * EMB + c];
    s += v;
    s2 += v * v;
  }
  atomicAdd(&sums[c], s);
  atomicAdd(&sumsq[c], s2);
}

// ---------------- BN apply (+optional ReLU) ----------------
__global__ void bn_apply_kernel(const float* __restrict__ h3,
                                const float* __restrict__ sums,
                                const float* __restrict__ sumsq,
                                const float* __restrict__ gamma,
                                const float* __restrict__ beta,
                                float* __restrict__ out,
                                int n_nodes, int do_relu) {
  int tid = blockIdx.x * blockDim.x + threadIdx.x;
  if (tid >= n_nodes * EMB) return;
  int c = tid & (EMB - 1);
  float invn = 1.f / (float)n_nodes;
  float mu  = sums[c] * invn;
  float var = sumsq[c] * invn - mu * mu;       // biased variance
  float isd = rsqrtf(var + BNEPS);
  float v = (h3[tid] - mu) * isd * gamma[c] + beta[c];
  if (do_relu) v = v > 0.f ? v : 0.f;
  out[tid] = v;
}

// ---------------- host orchestration ----------------
extern "C" void kernel_launch(void* const* d_in, const int* in_sizes, int n_in,
                              void* d_out, int out_size, void* d_ws, size_t ws_size,
                              hipStream_t stream) {
  const int*   x     = (const int*)d_in[0];
  const int*   ei    = (const int*)d_in[1];
  const int*   ea    = (const int*)d_in[2];
  const float* atom  = (const float*)d_in[3];
  const float* chir  = (const float*)d_in[4];
  const float* hyb   = (const float*)d_in[5];
  const float* e1    = (const float*)d_in[6];
  const float* e2    = (const float*)d_in[7];
  const float* W1    = (const float*)d_in[8];
  const float* b1    = (const float*)d_in[9];
  const float* W2    = (const float*)d_in[10];
  const float* b2    = (const float*)d_in[11];
  const float* gamma = (const float*)d_in[12];
  const float* beta  = (const float*)d_in[13];
  float* out = (float*)d_out;

  const int N = in_sizes[0] / 3;   // 50000
  const int E = in_sizes[2] / 2;   // 600000
  (void)n_in; (void)out_size; (void)ws_size;

  // carve workspace (all offsets 256B aligned)
  char* ws = (char*)d_ws;
  size_t off = 0;
  auto carve = [&](size_t bytes) -> char* {
    char* p = ws + off;
    off = (off + bytes + 255) & ~(size_t)255;
    return p;
  };
  float*    h    = (float*)   carve((size_t)N * EMB * sizeof(float));
  float*    agg  = (float*)   carve((size_t)N * EMB * sizeof(float));
  _Float16* h2a  = (_Float16*)carve((size_t)N * HID * sizeof(_Float16));
  float*    h3   = (float*)   carve((size_t)N * EMB * sizeof(float));
  _Float16* W1t  = (_Float16*)carve((size_t)L * EMB * HID * sizeof(_Float16));
  _Float16* W2t  = (_Float16*)carve((size_t)L * EMB * HID * sizeof(_Float16));
  float*    sums = (float*)   carve(EMB * sizeof(float));
  float*    sumq = (float*)   carve(EMB * sizeof(float));

  // input embedding
  {
    int total = N * EMB;
    node_embed_kernel<<<(total + 255) / 256, 256, 0, stream>>>(x, atom, chir, hyb, h, N);
  }
  // weight convert + transpose (all layers, once per call)
  {
    int total = 2 * L * EMB * HID;
    convert_weights_kernel<<<(total + 255) / 256, 256, 0, stream>>>(W1, W2, W1t, W2t);
  }

  const int mt = (N + 15) >> 4;
  for (int l = 0; l < L; ++l) {
    hipMemsetAsync(agg,  0, (size_t)N * EMB * sizeof(float), stream);
    hipMemsetAsync(sums, 0, EMB * sizeof(float), stream);
    hipMemsetAsync(sumq, 0, EMB * sizeof(float), stream);

    {
      long long total = (long long)E * 32;
      int grid = (int)((total + 255) / 256);
      edge_scatter_kernel<<<grid, 256, 0, stream>>>(
          h, ei, ea, e1 + (size_t)l * 6 * EMB, e2 + (size_t)l * 4 * EMB, agg, E);
    }
    {
      int tiles = mt * (HID / 64);       // waves (16x64 tiles)
      gemm1_relu_kernel<<<(tiles + 7) / 8, 256, 0, stream>>>(
          agg, W1t + (size_t)l * EMB * HID, b1 + (size_t)l * HID, h2a, N);
    }
    {
      int tiles = mt * (EMB / 64);
      gemm2_kernel<<<(tiles + 7) / 8, 256, 0, stream>>>(
          h2a, W2t + (size_t)l * EMB * HID, b2 + (size_t)l * EMB, h3, N);
    }
    bn_stats_kernel<<<256, EMB, 0, stream>>>(h3, sums, sumq, N);
    {
      int total = N * EMB;
      float* dst = (l == L - 1) ? out : h;
      bn_apply_kernel<<<(total + 255) / 256, 256, 0, stream>>>(
          h3, sums, sumq, gamma + (size_t)l * EMB, beta + (size_t)l * EMB,
          dst, N, (l != L - 1) ? 1 : 0);
    }
  }
}
